// TaylorIntegralAttention_12249246728990
// MI455X (gfx1250) — matrix operand order repaired and checked by reading, not compiled, not run on hardware
//
#include <hip/hip_runtime.h>
#include <hip/hip_bf16.h>

// ---------------------------------------------------------------------------
// Taylor Integral Attention on gfx1250 (MI455X), all GEMMs via
// v_wmma_f32_16x16x32_bf16. phi() features are generated on the fly inside
// the WMMA A-fragment construction; denom is fused as an extra "ones" column
// of V. fp32 accumulation everywhere. All LDS fragment traffic is 128-bit.
// Global->LDS staging uses GLOBAL_LOAD_ASYNC_TO_LDS_B128 when the toolchain
// exposes it (ASYNCcnt-tracked, bypasses VGPRs), else two-phase staged copy.
// ---------------------------------------------------------------------------

typedef __attribute__((ext_vector_type(16))) __bf16 v16bf;
typedef __attribute__((ext_vector_type(8)))  float  v8f;

struct alignas(16) U4 { unsigned int x, y, z, w; };   // POD 128-bit chunk

union ABFrag { v16bf v; U4 q[2]; unsigned int u[8]; };

#if defined(__gfx1250__) && __has_builtin(__builtin_amdgcn_global_load_async_to_lds_b128)
#define HAVE_ASYNC_LDS 1
typedef int v4i __attribute__((vector_size(16)));
typedef __attribute__((address_space(1))) v4i GV4;   // global 128-bit chunk
typedef __attribute__((address_space(3))) v4i LV4;   // LDS 128-bit chunk
__device__ __forceinline__ void async_cp16(const void* g, void* l) {
    __builtin_amdgcn_global_load_async_to_lds_b128((GV4*)g, (LV4*)l, 0, 0);
}
__device__ __forceinline__ void wait_async0() {
#if __has_builtin(__builtin_amdgcn_s_wait_asynccnt)
    __builtin_amdgcn_s_wait_asynccnt(0);
#else
    asm volatile("s_wait_asynccnt 0x0" ::: "memory");
#endif
}
#else
#define HAVE_ASYNC_LDS 0
#endif

__device__ __forceinline__ unsigned int pack_bf16(float a, float b) {
    __bf16 b0 = (__bf16)a, b1 = (__bf16)b;
    return (unsigned int)__builtin_bit_cast(unsigned short, b0) |
           ((unsigned int)__builtin_bit_cast(unsigned short, b1) << 16);
}
__device__ __forceinline__ float bf_lo(unsigned int u) {
    return (float)__builtin_bit_cast(__bf16, (unsigned short)(u & 0xffffu));
}
__device__ __forceinline__ float bf_hi(unsigned int u) {
    return (float)__builtin_bit_cast(__bf16, (unsigned short)(u >> 16));
}

// ---------------------------------------------------------------------------
// fp32 -> bf16 conversion
// ---------------------------------------------------------------------------
__global__ void cvt_bf16_kernel(const float* __restrict__ src,
                                __bf16* __restrict__ dst, int n) {
    int i = blockIdx.x * blockDim.x + threadIdx.x;
    if (i < n) dst[i] = (__bf16)src[i];
}

// ---------------------------------------------------------------------------
// Fill Vt padding rows: d==64 -> 1.0 (ones column => k_sum / denom),
// d in 65..79 -> 0.0
// ---------------------------------------------------------------------------
__global__ void vt_init_kernel(__bf16* __restrict__ vt) {
    int idx = blockIdx.x * blockDim.x + threadIdx.x;   // 32 * 16 * 1024
    if (idx >= 32 * 16 * 1024) return;
    int m  = idx & 1023;
    int d  = 64 + ((idx >> 10) & 15);
    int bh = idx >> 14;
    vt[((size_t)bh * 80 + d) * 1024 + m] = (__bf16)((d == 64) ? 1.0f : 0.0f);
}

// ---------------------------------------------------------------------------
// Tiled WMMA GEMM: out[m][n] = sum_k A[m][k] * W[n][k] (+ bias[n])
// 128 threads = 4 waves; block tile 64x64; each wave 32x32 (2x2 WMMA tiles).
// Epilogue modes:
//   0: XQ[g][ntok][i]  = (acc+bq)*qscale   (bf16, g=(b*16+h)*2+s)
//   1: kT[g][i][ntok]  =  acc+bk           (bf16, transposed)
//   2: Vt[bh][d][ntok] =  acc+bv           (bf16, transposed, d-stride 80)
//   3: outF[m][n]      =  acc+bo           (fp32 final output)
// ---------------------------------------------------------------------------
__global__ void gemm_proj_kernel(const __bf16* __restrict__ A,
                                 const __bf16* __restrict__ W,
                                 const float* __restrict__ bias,
                                 int M, int N, int K, int mode, float qscale,
                                 __bf16* __restrict__ outXQ,
                                 __bf16* __restrict__ outT,
                                 float* __restrict__ outF) {
    __shared__ __align__(16) __bf16 As[64][32];
    __shared__ __align__(16) __bf16 Bs[64][32];

    const int tid  = threadIdx.x;        // 0..127
    const int wave = tid >> 5;           // 0..3
    const int lane = tid & 31;
    const int wm   = (wave >> 1) * 32;   // wave M offset in block tile
    const int wn   = (wave & 1) * 32;    // wave N offset in block tile
    const int m0   = blockIdx.y * 64;
    const int n0   = blockIdx.x * 64;
    const int lrow = lane & 15;
    const int lhi  = lane >> 4;

    v8f acc[2][2] = {};

    for (int k0 = 0; k0 < K; k0 += 32) {
        __syncthreads();
        // stage 64x32 bf16 tiles (256 U4 chunks, 2 per thread each matrix)
#if HAVE_ASYNC_LDS
        for (int it = 0; it < 2; ++it) {
            int pos = it * 128 + tid;            // 0..255
            int r = pos >> 2, c8 = (pos & 3) * 8;
            async_cp16(&A[(size_t)(m0 + r) * K + k0 + c8], &As[r][c8]);
            async_cp16(&W[(size_t)(n0 + r) * K + k0 + c8], &Bs[r][c8]);
        }
        wait_async0();
#else
        U4 ra[2], rb[2];
        for (int it = 0; it < 2; ++it) {
            int pos = it * 128 + tid;
            int r = pos >> 2, c8 = (pos & 3) * 8;
            ra[it] = *(const U4*)&A[(size_t)(m0 + r) * K + k0 + c8];
            rb[it] = *(const U4*)&W[(size_t)(n0 + r) * K + k0 + c8];
        }
        for (int it = 0; it < 2; ++it) {
            int pos = it * 128 + tid;
            int r = pos >> 2, c8 = (pos & 3) * 8;
            *(U4*)&As[r][c8] = ra[it];
            *(U4*)&Bs[r][c8] = rb[it];
        }
#endif
        if (k0 + 32 < K) {                        // prefetch next K tile
            __builtin_prefetch(&A[(size_t)(m0 + (tid >> 1)) * K + k0 + 32], 0, 3);
            __builtin_prefetch(&W[(size_t)(n0 + (tid >> 1)) * K + k0 + 32], 0, 3);
        }
        __syncthreads();

        ABFrag af[2], bfr[2];
        for (int i = 0; i < 2; ++i) {
            int mrow = wm + i * 16 + lrow;
            int nrow = wn + i * 16 + lrow;
            af[i].q[0]  = *(const U4*)&As[mrow][lhi * 8];
            af[i].q[1]  = *(const U4*)&As[mrow][16 + lhi * 8];
            bfr[i].q[0] = *(const U4*)&Bs[nrow][lhi * 8];
            bfr[i].q[1] = *(const U4*)&Bs[nrow][16 + lhi * 8];
        }
        for (int i = 0; i < 2; ++i)
            for (int j = 0; j < 2; ++j)
                acc[i][j] = __builtin_amdgcn_wmma_f32_16x16x32_bf16(
                    false, af[i].v, false, bfr[j].v, (short)0, acc[i][j],
                    false, false);
    }

    // epilogue
    for (int i = 0; i < 2; ++i) {
        for (int j = 0; j < 2; ++j) {
            int gn = n0 + wn + j * 16 + lrow;             // output column
            float bval = bias[gn];
            for (int r = 0; r < 8; ++r) {
                int gm = m0 + wm + i * 16 + lhi * 8 + r;  // output row (token)
                float val = acc[i][j][r] + bval;
                int b = gm >> 10, nt = gm & 1023;
                if (mode == 0) {
                    int h = gn >> 6, s = (gn >> 5) & 1, ii = gn & 31;
                    int g = (b * 16 + h) * 2 + s;
                    outXQ[((size_t)g * 1024 + nt) * 32 + ii] =
                        (__bf16)(val * qscale);
                } else if (mode == 1) {
                    int h = gn >> 6, s = (gn >> 5) & 1, ii = gn & 31;
                    int g = (b * 16 + h) * 2 + s;
                    outT[((size_t)g * 32 + ii) * 1024 + nt] = (__bf16)val;
                } else if (mode == 2) {
                    int h = gn >> 6, d = gn & 63;
                    int bh = b * 16 + h;
                    outT[((size_t)bh * 80 + d) * 1024 + nt] = (__bf16)val;
                } else {
                    outF[(size_t)gm * N + gn] = val;
                }
            }
        }
    }
}

// ---------------------------------------------------------------------------
// KV(f,d) = sum_m phi_k(f,m) * V'(m,d)   per group g = (b,h,s)
// F padded to 1088: f==0 -> 1; f in [16,48) -> kT[f-16][m];
// f in [48,1072) -> kT[i][m]*kT[j][m]/sqrt2 (f-48 = i*32+j); else 0.
// One wave per (group, 16-row f-tile); output 16f x 80d, loop over m.
// Stored transposed: KVt[g][d][f] bf16 (f-contiguous, feeds next GEMM's B).
// ---------------------------------------------------------------------------
__global__ void kv_gemm_kernel(const __bf16* __restrict__ kt,
                               const __bf16* __restrict__ vt,
                               __bf16* __restrict__ kvt) {
    __shared__ __align__(16) __bf16 Ks[32][32];   // kT[i][m-block]
    __shared__ __align__(16) __bf16 Vs[80][32];   // Vt[d][m-block]

    const int g    = blockIdx.y;
    const int f0   = blockIdx.x * 16;
    const int bh   = g >> 1;
    const int lane = threadIdx.x;
    const int lrow = lane & 15;
    const int lhi  = lane >> 4;
    const float inv_s2 = 0.70710678118654752f;

    const __bf16* ktg = kt + (size_t)g  * 32 * 1024;
    const __bf16* vtg = vt + (size_t)bh * 80 * 1024;

    v8f acc[5] = {};
    const int f = f0 + lrow;   // this lane's A-matrix row
    const int fi = (f - 48) >> 5, fj = (f - 48) & 31;   // quad indices

    for (int m0 = 0; m0 < 1024; m0 += 32) {
        __syncthreads();
#if HAVE_ASYNC_LDS
        for (int it = 0; it < 4; ++it) {           // 32x32 kT tile, 128 U4
            int pos = it * 32 + lane;
            int r = pos >> 2, c8 = (pos & 3) * 8;
            async_cp16(&ktg[(size_t)r * 1024 + m0 + c8], &Ks[r][c8]);
        }
        for (int it = 0; it < 10; ++it) {          // 80x32 V tile, 320 U4
            int pos = it * 32 + lane;
            int r = pos >> 2, c8 = (pos & 3) * 8;
            async_cp16(&vtg[(size_t)r * 1024 + m0 + c8], &Vs[r][c8]);
        }
        wait_async0();
#else
        for (int it = 0; it < 4; ++it) {
            int pos = it * 32 + lane;
            int r = pos >> 2, c8 = (pos & 3) * 8;
            *(U4*)&Ks[r][c8] = *(const U4*)&ktg[(size_t)r * 1024 + m0 + c8];
        }
        for (int it = 0; it < 10; ++it) {
            int pos = it * 32 + lane;
            int r = pos >> 2, c8 = (pos & 3) * 8;
            *(U4*)&Vs[r][c8] = *(const U4*)&vtg[(size_t)r * 1024 + m0 + c8];
        }
#endif
        __syncthreads();

        ABFrag af;
        if (f < 16) {
            unsigned int c = (f == 0) ? 0x3F803F80u : 0u;
            for (int v = 0; v < 8; ++v) af.u[v] = c;
        } else if (f < 48) {
            af.q[0] = *(const U4*)&Ks[f - 16][lhi * 8];
            af.q[1] = *(const U4*)&Ks[f - 16][16 + lhi * 8];
        } else if (f < 1072) {
            ABFrag ri, rj;
            ri.q[0] = *(const U4*)&Ks[fi][lhi * 8];
            ri.q[1] = *(const U4*)&Ks[fi][16 + lhi * 8];
            rj.q[0] = *(const U4*)&Ks[fj][lhi * 8];
            rj.q[1] = *(const U4*)&Ks[fj][16 + lhi * 8];
            for (int v = 0; v < 8; ++v)
                af.u[v] = pack_bf16(bf_lo(ri.u[v]) * bf_lo(rj.u[v]) * inv_s2,
                                    bf_hi(ri.u[v]) * bf_hi(rj.u[v]) * inv_s2);
        } else {
            for (int v = 0; v < 8; ++v) af.u[v] = 0u;
        }

        for (int dt = 0; dt < 5; ++dt) {
            ABFrag bfr;
            bfr.q[0] = *(const U4*)&Vs[dt * 16 + lrow][lhi * 8];
            bfr.q[1] = *(const U4*)&Vs[dt * 16 + lrow][16 + lhi * 8];
            acc[dt] = __builtin_amdgcn_wmma_f32_16x16x32_bf16(
                false, af.v, false, bfr.v, (short)0, acc[dt], false, false);
        }
    }

    for (int dt = 0; dt < 5; ++dt) {
        int d = dt * 16 + lrow;
        for (int r = 0; r < 8; ++r) {
            int fr = f0 + lhi * 8 + r;
            kvt[((size_t)g * 80 + d) * 1088 + fr] = (__bf16)acc[dt][r];
        }
    }
}

// ---------------------------------------------------------------------------
// numer(n,d) = sum_f phi_q(n,f) * KV(f,d)  per group; d==64 column is denom.
// One wave per (group, 16-row n-tile); loop over f (K dim, 1088).
// phi_q built on the fly from XQ (pre-scaled q) in LDS.
// ---------------------------------------------------------------------------
__global__ void numer_gemm_kernel(const __bf16* __restrict__ xq,
                                  const __bf16* __restrict__ kvt,
                                  float* __restrict__ numer) {
    __shared__ __align__(16) __bf16 Qs[16][32];   // xq[n-tile][i]
    __shared__ __align__(16) __bf16 Bs[80][32];   // KVt[d][f-block]

    const int g    = blockIdx.y;
    const int n0   = blockIdx.x * 16;
    const int lane = threadIdx.x;
    const int lrow = lane & 15;
    const int lhi  = lane >> 4;
    const float inv_s2 = 0.70710678118654752f;

    const __bf16* xqg = xq  + (size_t)g * 1024 * 32;
    const __bf16* kvg = kvt + (size_t)g * 80 * 1088;

    v8f acc[5] = {};

    for (int it = 0; it < 2; ++it) {              // 16x32 xq tile, 64 U4
        int pos = it * 32 + lane;
        int r = pos >> 2, c8 = (pos & 3) * 8;
        *(U4*)&Qs[r][c8] = *(const U4*)&xqg[(size_t)(n0 + r) * 32 + c8];
    }

    for (int f0 = 0; f0 < 1088; f0 += 32) {
        __syncthreads();
#if HAVE_ASYNC_LDS
        for (int it = 0; it < 10; ++it) {         // 80x32 KVt tile, 320 U4
            int pos = it * 32 + lane;
            int r = pos >> 2, c8 = (pos & 3) * 8;
            async_cp16(&kvg[(size_t)r * 1088 + f0 + c8], &Bs[r][c8]);
        }
        wait_async0();
#else
        for (int it = 0; it < 10; ++it) {
            int pos = it * 32 + lane;
            int r = pos >> 2, c8 = (pos & 3) * 8;
            *(U4*)&Bs[r][c8] = *(const U4*)&kvg[(size_t)r * 1088 + f0 + c8];
        }
#endif
        __syncthreads();

        ABFrag af;
        for (int v = 0; v < 8; ++v) {
            int fl = (v >> 2) * 16 + lhi * 8 + (v & 3) * 2;
            int fidx = f0 + fl;                   // even; pairs stay in-class
            unsigned int pair;
            if (fidx < 16) {
                pair = (fidx == 0) ? 0x3F803F80u : 0u;
            } else if (fidx < 48) {
                pair = *(const unsigned int*)&Qs[lrow][fidx - 16];
            } else if (fidx < 1072) {
                int fi = (fidx - 48) >> 5, fj = (fidx - 48) & 31;
                float qi = (float)Qs[lrow][fi];
                unsigned int qj = *(const unsigned int*)&Qs[lrow][fj];
                pair = pack_bf16(qi * bf_lo(qj) * inv_s2,
                                 qi * bf_hi(qj) * inv_s2);
            } else {
                pair = 0u;
            }
            af.u[v] = pair;
        }

        for (int dt = 0; dt < 5; ++dt) {
            ABFrag bfr;
            bfr.q[0] = *(const U4*)&Bs[dt * 16 + lrow][lhi * 8];
            bfr.q[1] = *(const U4*)&Bs[dt * 16 + lrow][16 + lhi * 8];
            acc[dt] = __builtin_amdgcn_wmma_f32_16x16x32_bf16(
                false, af.v, false, bfr.v, (short)0, acc[dt], false, false);
        }
    }

    for (int dt = 0; dt < 5; ++dt) {
        int d = dt * 16 + lrow;
        for (int r = 0; r < 8; ++r) {
            int n = n0 + lhi * 8 + r;
            numer[((size_t)g * 1024 + n) * 80 + d] = acc[dt][r];
        }
    }
}

// ---------------------------------------------------------------------------
// attn[m][h*64+d] = 0.5 * sum_s numer[g][n][d] / (numer[g][n][64] + 1e-6)
// ---------------------------------------------------------------------------
__global__ void combine_kernel(const float* __restrict__ numer,
                               __bf16* __restrict__ attn) {
    int idx = blockIdx.x * blockDim.x + threadIdx.x;   // 2048*1024
    if (idx >= 2048 * 1024) return;
    int c = idx & 1023, m = idx >> 10;
    int b = m >> 10, nt = m & 1023;
    int h = c >> 6, d = c & 63;
    int g0 = (b * 16 + h) * 2;
    float sum = 0.f;
    for (int s = 0; s < 2; ++s) {
        const float* base = numer + ((size_t)(g0 + s) * 1024 + nt) * 80;
        sum += base[d] / (base[64] + 1e-6f);
    }
    attn[idx] = (__bf16)(0.5f * sum);
}

// ---------------------------------------------------------------------------
extern "C" void kernel_launch(void* const* d_in, const int* in_sizes, int n_in,
                              void* d_out, int out_size, void* d_ws, size_t ws_size,
                              hipStream_t stream) {
    const float* query = (const float*)d_in[0];
    const float* key_p = (const float*)d_in[1];
    const float* value = (const float*)d_in[2];
    const float* Wq = (const float*)d_in[3];
    const float* bq = (const float*)d_in[4];
    const float* Wk = (const float*)d_in[5];
    const float* bk = (const float*)d_in[6];
    const float* Wv = (const float*)d_in[7];
    const float* bv = (const float*)d_in[8];
    const float* Wo = (const float*)d_in[9];
    const float* bo = (const float*)d_in[10];

    // workspace layout (bytes)
    const size_t E_TOK = 2048u * 1024u;     // token-matrix elems
    const size_t E_W   = 1024u * 1024u;     // weight elems
    char* ws = (char*)d_ws;
    size_t off = 0;
    __bf16* qb  = (__bf16*)(ws + off); off += E_TOK * 2;
    __bf16* kb  = (__bf16*)(ws + off); off += E_TOK * 2;
    __bf16* vb  = (__bf16*)(ws + off); off += E_TOK * 2;
    __bf16* wqb = (__bf16*)(ws + off); off += E_W * 2;
    __bf16* wkb = (__bf16*)(ws + off); off += E_W * 2;
    __bf16* wvb = (__bf16*)(ws + off); off += E_W * 2;
    __bf16* wob = (__bf16*)(ws + off); off += E_W * 2;
    __bf16* XQ  = (__bf16*)(ws + off); off += (size_t)64 * 1024 * 32 * 2;
    __bf16* KT  = (__bf16*)(ws + off); off += (size_t)64 * 32 * 1024 * 2;
    __bf16* VT  = (__bf16*)(ws + off); off += (size_t)32 * 80 * 1024 * 2;
    __bf16* KVT = (__bf16*)(ws + off); off += (size_t)64 * 80 * 1088 * 2;
    float*  NUM = (float*) (ws + off); off += (size_t)64 * 1024 * 80 * 4;
    __bf16* ATT = (__bf16*)(ws + off); off += E_TOK * 2;

    const int ntok = (int)E_TOK, nw = (int)E_W;
    cvt_bf16_kernel<<<(ntok + 255) / 256, 256, 0, stream>>>(query, qb, ntok);
    cvt_bf16_kernel<<<(ntok + 255) / 256, 256, 0, stream>>>(key_p, kb, ntok);
    cvt_bf16_kernel<<<(ntok + 255) / 256, 256, 0, stream>>>(value, vb, ntok);
    cvt_bf16_kernel<<<(nw + 255) / 256, 256, 0, stream>>>(Wq, wqb, nw);
    cvt_bf16_kernel<<<(nw + 255) / 256, 256, 0, stream>>>(Wk, wkb, nw);
    cvt_bf16_kernel<<<(nw + 255) / 256, 256, 0, stream>>>(Wv, wvb, nw);
    cvt_bf16_kernel<<<(nw + 255) / 256, 256, 0, stream>>>(Wo, wob, nw);
    vt_init_kernel<<<(32 * 16 * 1024 + 255) / 256, 256, 0, stream>>>(VT);

    const float qscale = 0.17677669529663687f;   // 32^-0.5
    dim3 pg(16, 32);                              // N/64 x M/64
    gemm_proj_kernel<<<pg, 128, 0, stream>>>(qb, wqb, bq, 2048, 1024, 1024,
                                             0, qscale, XQ, nullptr, nullptr);
    gemm_proj_kernel<<<pg, 128, 0, stream>>>(kb, wkb, bk, 2048, 1024, 1024,
                                             1, 1.f, nullptr, KT, nullptr);
    gemm_proj_kernel<<<pg, 128, 0, stream>>>(vb, wvb, bv, 2048, 1024, 1024,
                                             2, 1.f, nullptr, VT, nullptr);

    kv_gemm_kernel<<<dim3(68, 64), 32, 0, stream>>>(KT, VT, KVT);
    numer_gemm_kernel<<<dim3(64, 64), 32, 0, stream>>>(XQ, KVT, NUM);
    combine_kernel<<<(2048 * 1024 + 255) / 256, 256, 0, stream>>>(NUM, ATT);

    gemm_proj_kernel<<<pg, 128, 0, stream>>>(ATT, wob, bo, 2048, 1024, 1024,
                                             3, 1.f, nullptr, nullptr,
                                             (float*)d_out);
    (void)in_sizes; (void)n_in; (void)out_size; (void)ws_size;
}
